// BiLSTMSMNIST_20538533609764
// MI455X (gfx1250) — matrix operand order repaired and checked by reading, not compiled, not run on hardware
//
#include <hip/hip_runtime.h>
#include <hip/hip_bf16.h>

// ---------------- problem constants ----------------
#define T_STEPS 784
#define BATCH   512
#define HID     200
#define HID4    800
#define KPAD    224          // HID padded to multiple of 32 for WMMA K
#define NTILES  50           // 800 / 16
#define KCHUNKS 7            // 224 / 32
#define ROWS_PER_WG 16       // one WMMA M-tile of batch rows per workgroup
#define WGS_PER_DIR 32       // 512 / 16
#define PACK_HALVES_PER_DIR (KCHUNKS * NTILES * 32 * 16)   // 179200 halves

typedef _Float16 v16h __attribute__((ext_vector_type(16)));
typedef _Float16 v8h  __attribute__((ext_vector_type(8)));
typedef float    v8f  __attribute__((ext_vector_type(8)));

union ABPack { v16h v; v8h h[2]; };

__device__ __forceinline__ float sigm(float x) { return 1.0f / (1.0f + __expf(-x)); }

// ---------------------------------------------------------------------------
// Kernel 1: rank-1 collapse of the input-side GEMMs (D == 1).
//   u[j] = Wih[j,:] . W_in[:,0]      v[j] = Wih[j,:] . b_in + b[j]
// so g_in[t,b,j] = x[t,b]*u[j] + v[j].
// ---------------------------------------------------------------------------
__global__ void prep_uv(const float* __restrict__ W_in, const float* __restrict__ b_in,
                        const float* __restrict__ Wih_f, const float* __restrict__ b_f,
                        const float* __restrict__ Wih_b, const float* __restrict__ b_b,
                        float* __restrict__ U, float* __restrict__ V) {
  int gid = blockIdx.x * blockDim.x + threadIdx.x;
  if (gid >= 2 * HID4) return;
  int d = gid / HID4, j = gid - d * HID4;
  const float* Wih = d ? Wih_b : Wih_f;
  const float* bb  = d ? b_b   : b_f;
  float u = 0.f, v = 0.f;
  for (int k = 0; k < HID; ++k) {
    float w = Wih[j * HID + k];
    u += w * W_in[k];
    v += w * b_in[k];
  }
  U[gid] = u;
  V[gid] = v + bb[j];
}

// ---------------------------------------------------------------------------
// Kernel 2: pre-swizzle Whh^T (f32 -> f16) into the CDNA5 WMMA B-operand
// lane layout for V_WMMA_F32_16X16X32_F16:
//   lane l holds column n = nt*16 + (l&15); its 16 halves cover
//   k = kc*32 + (e>>3)*16 + (l>>4)*8 + (e&7)  (CDNA5 ISA, 16-bit B layout).
// Each lane's operand becomes one contiguous 32-byte global read.
// ---------------------------------------------------------------------------
__global__ void pack_whh(const float* __restrict__ Whh_f, const float* __restrict__ Whh_b,
                         _Float16* __restrict__ Wpack) {
  int gid = blockIdx.x * blockDim.x + threadIdx.x;
  const int total = 2 * KCHUNKS * NTILES * 32;
  if (gid >= total) return;
  int l = gid & 31;
  int r = gid >> 5;
  int nt = r % NTILES; r /= NTILES;
  int kc = r % KCHUNKS;
  int d  = r / KCHUNKS;
  const float* Whh = d ? Whh_b : Whh_f;     // (4H, H) row-major
  int n = nt * 16 + (l & 15);               // output column j of g = h @ Whh^T
  _Float16* dst = Wpack + (size_t)d * PACK_HALVES_PER_DIR
                + ((size_t)((kc * NTILES + nt) * 32 + l)) * 16;
#pragma unroll
  for (int e = 0; e < 16; ++e) {
    int k = kc * 32 + ((e >> 3) * 16) + ((l >> 4) * 8) + (e & 7);
    float val = (k < HID) ? Whh[n * HID + k] : 0.0f;
    dst[e] = (_Float16)val;
  }
}

// ---------------------------------------------------------------------------
// Kernel 3: bidirectional LSTM scan. 64 independent workgroups (batch rows
// never couple), each owning 16 batch rows of one direction.
// Per step: 16x800x224 GEMM via v_wmma_f32_16x16x32_f16, software-pipelined
// (loads for kc+1 issued before the WMMAs consuming kc) and two N-tiles per
// wave iteration sharing one A operand; then the pointwise cell update; then
// a deterministic per-WG batch partial sum of h.
// ---------------------------------------------------------------------------
__global__ void __launch_bounds__(256) bilstm_scan(
    const float* __restrict__ x,
    const float* __restrict__ U, const float* __restrict__ V,
    const _Float16* __restrict__ Wpack,
    float* __restrict__ hpart) {
  extern __shared__ char smem[];
  _Float16* h_lds = (_Float16*)smem;                 // [16][224] f16, zero-padded K
  float*    c_lds = (float*)(smem + 7168);           // [16][200] f32
  float*    g_lds = (float*)(smem + 19968);          // [16][800] f32 gate pre-acts
  float*    u_lds = (float*)(smem + 71168);          // [800]
  float*    v_lds = (float*)(smem + 74368);          // [800]
  float*    x_lds = (float*)(smem + 77568);          // [16]

  const int tid   = threadIdx.x;
  const int lane  = tid & 31;
  const int wave  = tid >> 5;
  const int wg    = blockIdx.x;
  const int dir   = wg >> 5;               // 0 = forward, 1 = backward
  const int wgIdx = wg & 31;
  const int m0    = wgIdx * ROWS_PER_WG;   // first batch row owned
  const _Float16* wp = Wpack + (size_t)dir * PACK_HALVES_PER_DIR;

  for (int i = tid; i < ROWS_PER_WG * KPAD; i += 256) h_lds[i] = (_Float16)0.0f;
  for (int i = tid; i < ROWS_PER_WG * HID;  i += 256) c_lds[i] = 0.0f;
  for (int i = tid; i < HID4; i += 256) {
    u_lds[i] = U[dir * HID4 + i];
    v_lds[i] = V[dir * HID4 + i];
  }
  __syncthreads();

  // A-operand LDS base (ISA 16-bit A layout: row = lane&15,
  // halves e=0..7 -> k0..k0+7, e=8..15 -> k0+16..k0+23, k0 = (lane>>4)*8).
  const int aRowOff = (lane & 15) * KPAD + ((lane >> 4) * 8);
  // C/D layout: lane l, reg r -> (m = (l>>4)*8 + r, n = l&15)
  const int gm  = (lane >> 4) * 8;
  const int gn0 = lane & 15;

  for (int s = 0; s < T_STEPS; ++s) {
    const int t = dir ? (T_STEPS - 1 - s) : s;
    if (tid < ROWS_PER_WG) x_lds[tid] = x[t * BATCH + m0 + tid];

    // ---- recurrent GEMM: G[16 x 800] = h[16 x 224] * WhhT ----
    // pairs of N-tiles (p, p+1): p = 2*wave + 16*i covers tiles 0..47
    for (int p = 2 * wave; p < 48; p += 16) {
      v8f acc0 = {}, acc1 = {};
      ABPack aC, b0C, b1C, aN, b0N, b1N;
      {
        const _Float16* ap = h_lds + aRowOff;
        aC.h[0] = *(const v8h*)ap;  aC.h[1] = *(const v8h*)(ap + 16);
        const _Float16* bp0 = wp + ((size_t)(p * 32 + lane)) * 16;
        b0C.h[0] = *(const v8h*)bp0; b0C.h[1] = *(const v8h*)(bp0 + 8);
        const _Float16* bp1 = bp0 + 32 * 16;   // tile p+1, same kc
        b1C.h[0] = *(const v8h*)bp1; b1C.h[1] = *(const v8h*)(bp1 + 8);
      }
#pragma unroll
      for (int kc = 0; kc < KCHUNKS; ++kc) {
        if (kc + 1 < KCHUNKS) {    // prefetch next K-chunk operands
          const _Float16* ap = h_lds + aRowOff + (kc + 1) * 32;
          aN.h[0] = *(const v8h*)ap;  aN.h[1] = *(const v8h*)(ap + 16);
          const _Float16* bp0 = wp + ((size_t)(((kc + 1) * NTILES + p) * 32 + lane)) * 16;
          b0N.h[0] = *(const v8h*)bp0; b0N.h[1] = *(const v8h*)(bp0 + 8);
          const _Float16* bp1 = bp0 + 32 * 16;
          b1N.h[0] = *(const v8h*)bp1; b1N.h[1] = *(const v8h*)(bp1 + 8);
        }
        acc0 = __builtin_amdgcn_wmma_f32_16x16x32_f16(false, aC.v, false, b0C.v,
                                                      (short)0, acc0, false, false);
        acc1 = __builtin_amdgcn_wmma_f32_16x16x32_f16(false, aC.v, false, b1C.v,
                                                      (short)0, acc1, false, false);
        aC = aN; b0C = b0N; b1C = b1N;
      }
#pragma unroll
      for (int r = 0; r < 8; ++r) {
        g_lds[(gm + r) * HID4 + p * 16 + gn0]       = acc0[r];
        g_lds[(gm + r) * HID4 + (p + 1) * 16 + gn0] = acc1[r];
      }
    }
    // remainder tiles 48, 49 -> waves 0, 1
    if (wave < 2) {
      const int nt = 48 + wave;
      v8f acc = {};
      ABPack aC, bC, aN, bN;
      {
        const _Float16* ap = h_lds + aRowOff;
        aC.h[0] = *(const v8h*)ap;  aC.h[1] = *(const v8h*)(ap + 16);
        const _Float16* bp = wp + ((size_t)(nt * 32 + lane)) * 16;
        bC.h[0] = *(const v8h*)bp;  bC.h[1] = *(const v8h*)(bp + 8);
      }
#pragma unroll
      for (int kc = 0; kc < KCHUNKS; ++kc) {
        if (kc + 1 < KCHUNKS) {
          const _Float16* ap = h_lds + aRowOff + (kc + 1) * 32;
          aN.h[0] = *(const v8h*)ap;  aN.h[1] = *(const v8h*)(ap + 16);
          const _Float16* bp = wp + ((size_t)(((kc + 1) * NTILES + nt) * 32 + lane)) * 16;
          bN.h[0] = *(const v8h*)bp;  bN.h[1] = *(const v8h*)(bp + 8);
        }
        acc = __builtin_amdgcn_wmma_f32_16x16x32_f16(false, aC.v, false, bC.v,
                                                     (short)0, acc, false, false);
        aC = aN; bC = bN;
      }
#pragma unroll
      for (int r = 0; r < 8; ++r)
        g_lds[(gm + r) * HID4 + nt * 16 + gn0] = acc[r];
    }
    __syncthreads();

    // ---- pointwise LSTM cell (gate order i, f, g, o) ----
    for (int e = tid; e < ROWS_PER_WG * HID; e += 256) {
      int m  = e / HID;
      int hx = e - m * HID;
      float xv = x_lds[m];
      float gi = g_lds[m * HID4 +       hx] + xv * u_lds[      hx] + v_lds[      hx];
      float gf = g_lds[m * HID4 + 200 + hx] + xv * u_lds[200 + hx] + v_lds[200 + hx];
      float gg = g_lds[m * HID4 + 400 + hx] + xv * u_lds[400 + hx] + v_lds[400 + hx];
      float go = g_lds[m * HID4 + 600 + hx] + xv * u_lds[600 + hx] + v_lds[600 + hx];
      float c = c_lds[m * HID + hx];
      c = sigm(gf) * c + sigm(gi) * tanhf(gg);
      float h = sigm(go) * tanhf(c);
      c_lds[m * HID + hx] = c;
      h_lds[m * KPAD + hx] = (_Float16)h;
    }
    __syncthreads();

    // ---- deterministic per-WG batch partial sum of h for the mean ----
    if (tid < HID) {
      float ssum = 0.f;
#pragma unroll
      for (int m = 0; m < ROWS_PER_WG; ++m) ssum += (float)h_lds[m * KPAD + tid];
      hpart[((size_t)(dir * WGS_PER_DIR + wgIdx) * T_STEPS + t) * HID + tid] = ssum;
    }
    // no barrier needed here: next iter only reads h_lds before the next
    // __syncthreads(), and writes to it only after it.
  }
}

// ---------------------------------------------------------------------------
// Kernel 4: reduce per-WG partials -> hbar[t] (mean over batch), then the
// tiny (T,400)x(400,10) projection.
// ---------------------------------------------------------------------------
__global__ void finalize_out(const float* __restrict__ hpart,
                             const float* __restrict__ W_fc,
                             const float* __restrict__ b_fc,
                             float* __restrict__ out) {
  __shared__ float hbar[2 * HID];
  const int t = blockIdx.x;
  const int tid = threadIdx.x;
  for (int q = tid; q < 2 * HID; q += 256) {
    int d = q / HID, hx = q - d * HID;
    float s = 0.f;
    for (int w = 0; w < WGS_PER_DIR; ++w)
      s += hpart[((size_t)(d * WGS_PER_DIR + w) * T_STEPS + t) * HID + hx];
    hbar[q] = s * (1.0f / (float)BATCH);
  }
  __syncthreads();
  if (tid < 10) {
    float s = b_fc[tid];
    for (int q = 0; q < 2 * HID; ++q) s += hbar[q] * W_fc[tid * (2 * HID) + q];
    out[t * 10 + tid] = s;
  }
}

// ---------------------------------------------------------------------------
extern "C" void kernel_launch(void* const* d_in, const int* in_sizes, int n_in,
                              void* d_out, int out_size, void* d_ws, size_t ws_size,
                              hipStream_t stream) {
  (void)in_sizes; (void)n_in; (void)out_size; (void)ws_size;
  const float* x     = (const float*)d_in[0];
  const float* W_in  = (const float*)d_in[1];
  const float* b_in  = (const float*)d_in[2];
  const float* Wih_f = (const float*)d_in[3];
  const float* Whh_f = (const float*)d_in[4];
  const float* b_f   = (const float*)d_in[5];
  const float* Wih_b = (const float*)d_in[6];
  const float* Whh_b = (const float*)d_in[7];
  const float* b_b   = (const float*)d_in[8];
  const float* W_fc  = (const float*)d_in[9];
  const float* b_fc  = (const float*)d_in[10];
  float* out = (float*)d_out;

  // workspace layout (all offsets 256B aligned)
  char* ws = (char*)d_ws;
  float*    U     = (float*)(ws + 0);        //  2*800 f32   =   6400 B
  float*    V     = (float*)(ws + 6400);     //  2*800 f32   =   6400 B
  _Float16* Wpack = (_Float16*)(ws + 12800); //  2*179200 f16 = 716800 B
  float*    hpart = (float*)(ws + 729600);   //  2*32*784*200 f32 = 40.1 MB

  prep_uv<<<7, 256, 0, stream>>>(W_in, b_in, Wih_f, b_f, Wih_b, b_b, U, V);
  pack_whh<<<88, 256, 0, stream>>>(Whh_f, Whh_b, Wpack);
  bilstm_scan<<<2 * WGS_PER_DIR, 256, 77632, stream>>>(x, U, V, Wpack, hpart);
  finalize_out<<<T_STEPS, 256, 0, stream>>>(hpart, W_fc, b_fc, out);
}
